// QuantizeEMA_51410758533674
// MI455X (gfx1250) — matrix operand order repaired and checked by reading, not compiled, not run on hardware
//
#include <hip/hip_runtime.h>
#include <hip/hip_bf16.h>

typedef __attribute__((ext_vector_type(16))) __bf16 v16bf;
typedef __attribute__((ext_vector_type(8)))  float  v8f;
typedef __attribute__((ext_vector_type(4)))  unsigned int u32x4;
typedef __attribute__((ext_vector_type(8)))  int i32x8;
typedef __attribute__((ext_vector_type(4)))  int i32x4;

#define DECAY 0.99f
#define ONE_MINUS_DECAY 0.01f
#define EPSV 1e-5f

// ---- sizes ----
// B=64 C=256 H=W=32 -> HW=1024, N=65536, n_embed=1024
#define NQ   16777216   // quantize_out elements (B*C*H*W)
#define NDF  16777216   // diff elements (B*H*W*C)
#define NI   65536      // embed_ind elements
#define NE2  262144     // C*n_embed

// bf16 helpers via bit ops (round-to-nearest-even)
__device__ __forceinline__ __bf16 f2bf(float f) {
    unsigned u = __builtin_bit_cast(unsigned, f);
    unsigned r = (u + 0x7fffu + ((u >> 16) & 1u)) >> 16;
    unsigned short s = (unsigned short)r;
    return __builtin_bit_cast(__bf16, s);
}
__device__ __forceinline__ float bf2f(__bf16 h) {
    unsigned short s = __builtin_bit_cast(unsigned short, h);
    unsigned u = ((unsigned)s) << 16;
    return __builtin_bit_cast(float, u);
}

// Issue one TDM copy of 16KB (2048 x 8B) from global to LDS. 1-D tile.
// 6-arg builtin form: (g0 u32x4, g1 i32x8, g2 i32x4, g3 i32x4, i32x8, cpol)
__device__ __forceinline__ void tdm_copy16k(const __bf16* gsrc, unsigned lds_off) {
    unsigned long long ga = (unsigned long long)(uintptr_t)gsrc;
    u32x4 g0;
    g0[0] = 1u;                                              // count=1, user mode
    g0[1] = lds_off;                                         // lds_addr (bytes)
    g0[2] = (unsigned)(ga & 0xffffffffu);                    // global_addr[31:0]
    g0[3] = (unsigned)((ga >> 32) & 0x01ffffffu) | (2u << 30); // addr[56:32] | type=2
    i32x8 g1;
    g1[0] = 0x00030000;               // wg_mask=0, data_size=3 (8B)
    g1[1] = (int)(2048u << 16);       // tensor_dim0 = 2048 (low16)
    g1[2] = (int)(1u << 16);          // tensor_dim1 = 1
    g1[3] = (int)(2048u << 16);       // tile_dim0 = 2048
    g1[4] = 1;                        // tile_dim1 = 1, tile_dim2 = 0
    g1[5] = 2048;                     // tensor_dim0_stride low32
    g1[6] = 0;
    g1[7] = 0;
    i32x4 z4 = {0, 0, 0, 0};
    i32x8 z8 = {0, 0, 0, 0, 0, 0, 0, 0};
    __builtin_amdgcn_tensor_load_to_lds(g0, g1, z4, z4, z8, 0);
}

// ---------------------------------------------------------------------------
// Zero scratch accumulators (onehot + embed_sum are contiguous)
__global__ void k_zero(float* __restrict__ p, int n) {
    int i = blockIdx.x * blockDim.x + threadIdx.x;
    if (i < n) p[i] = 0.0f;
}

// ---------------------------------------------------------------------------
// Pre-swizzle embed (C x 1024 fp32) into bf16 hi/lo B-fragments, packed per
// N-tile: ehl[jn*8192 + kt*512 + L*16 + e] (hi), +4096 (lo) -> one contiguous
// 16KB block per jn, TDM-copyable in a single descriptor.
// B 32x16 layout: N = L%16; lanes 0-15 hold K=0..15 (reg r -> K=2r,2r+1),
// lanes 16-31 hold K=16..31.
__global__ void k_prep_embed(const float* __restrict__ embed,
                             __bf16* __restrict__ ehl) {
    int g = blockIdx.x * blockDim.x + threadIdx.x;      // 0..16383
    int L  = g & 31;
    int kt = (g >> 5) & 7;
    int jn = g >> 8;                                    // 0..63
    int code  = (jn << 4) + (L & 15);
    int kbase = kt * 32 + ((L < 16) ? 0 : 16);
    size_t off = (size_t)jn * 8192 + (size_t)kt * 512 + (size_t)L * 16;
    for (int r = 0; r < 8; ++r)
        for (int p = 0; p < 2; ++p) {
            int c = kbase + 2 * r + p;
            float f = embed[(size_t)c * 1024 + code];
            __bf16 h = f2bf(f);
            ehl[off + 2 * r + p]        = h;
            ehl[off + 4096 + 2 * r + p] = f2bf(f - bf2f(h));
        }
}

// ||e_j||^2 (fp32, exact)
__global__ void k_enorm(const float* __restrict__ embed, float* __restrict__ enorm) {
    int j = blockIdx.x * blockDim.x + threadIdx.x;      // 0..1023
    if (j >= 1024) return;
    float s = 0.0f;
    for (int c = 0; c < 256; ++c) {
        float v = embed[(size_t)c * 1024 + j];
        s += v * v;
    }
    enorm[j] = s;
}

// ---------------------------------------------------------------------------
// Fused distance GEMM (bf16x3 error-compensated) + argmin.
// 4 waves/block, each owns a 16-row M tile; B panel double-buffered in LDS,
// staged by the Tensor Data Mover (wave 0 issues, TENSORcnt + barrier sync).
__global__ __launch_bounds__(128) void k_dist_argmin(
    const float* __restrict__ inp, const __bf16* __restrict__ ehl,
    const float* __restrict__ enorm, int* __restrict__ ind,
    float* __restrict__ ind_f)
{
    __shared__ __align__(128) __bf16 sB[2][8192];   // 2 x 16KB double buffer
    int blk  = blockIdx.x;                 // 0..1023 (64-row super-tiles)
    int wave = threadIdx.x >> 5;
    int lane = threadIdx.x & 31;
    int Mloc = lane & 15;
    int klo  = (lane < 16) ? 0 : 16;
    int b     = blk >> 4;                  // 16 blocks of 64 rows per image b
    int hwrow = ((blk & 15) << 6) + (wave << 4) + Mloc;
    const float* rowp = inp + ((size_t)b << 18) + hwrow;   // + b*256*1024

    unsigned lds0 = (unsigned)(uintptr_t)&sB[0][0];
    unsigned lds1 = (unsigned)(uintptr_t)&sB[1][0];

    // Prologue: TDM stages jn=0 panel into buffer 0 while we build A frags.
    if (wave == 0) tdm_copy16k(ehl, lds0);

    // A 16x32 bf16 hi/lo fragments, resident (128 VGPRs).
    v16bf Ah[8], Al[8];
#pragma unroll
    for (int kt = 0; kt < 8; ++kt) {
        int kb = kt * 32 + klo;
#pragma unroll
        for (int v = 0; v < 8; ++v) {
            int k = kb + 2 * v;
            float f0 = rowp[(size_t)k * 1024];
            float f1 = rowp[(size_t)(k + 1) * 1024];
            __bf16 h0 = f2bf(f0), h1 = f2bf(f1);
            Ah[kt][2 * v]     = h0;
            Ah[kt][2 * v + 1] = h1;
            Al[kt][2 * v]     = f2bf(f0 - bf2f(h0));
            Al[kt][2 * v + 1] = f2bf(f1 - bf2f(h1));
        }
    }

    float mval[8]; int midx[8];
#pragma unroll
    for (int i = 0; i < 8; ++i) { mval[i] = 3.4028235e38f; midx[i] = 0; }

    for (int jn = 0; jn < 64; ++jn) {
        if (wave == 0) {
            if (jn < 63) {
                // prefetch next panel into the other buffer, then wait for cur
                tdm_copy16k(ehl + (size_t)(jn + 1) * 8192, (jn & 1) ? lds0 : lds1);
                __builtin_amdgcn_s_wait_tensorcnt(1);
            } else {
                __builtin_amdgcn_s_wait_tensorcnt(0);
            }
        }
        __syncthreads();                       // publish buffer to all 4 waves

        const __bf16* buf = (jn & 1) ? &sB[1][0] : &sB[0][0];
        v8f acc0 = {0.f, 0.f, 0.f, 0.f, 0.f, 0.f, 0.f, 0.f};
        v8f acc1 = acc0;
#pragma unroll
        for (int kt = 0; kt < 8; ++kt) {
            v16bf Bh = *(const v16bf*)(buf + kt * 512 + lane * 16);
            v16bf Bl = *(const v16bf*)(buf + 4096 + kt * 512 + lane * 16);
            if (kt & 1) {  // two independent accumulation chains for ILP
                acc1 = __builtin_amdgcn_wmma_f32_16x16x32_bf16(false, Al[kt], false, Bh, (short)0, acc1, false, false);
                acc1 = __builtin_amdgcn_wmma_f32_16x16x32_bf16(false, Ah[kt], false, Bl, (short)0, acc1, false, false);
                acc1 = __builtin_amdgcn_wmma_f32_16x16x32_bf16(false, Ah[kt], false, Bh, (short)0, acc1, false, false);
            } else {
                acc0 = __builtin_amdgcn_wmma_f32_16x16x32_bf16(false, Al[kt], false, Bh, (short)0, acc0, false, false);
                acc0 = __builtin_amdgcn_wmma_f32_16x16x32_bf16(false, Ah[kt], false, Bl, (short)0, acc0, false, false);
                acc0 = __builtin_amdgcn_wmma_f32_16x16x32_bf16(false, Ah[kt], false, Bh, (short)0, acc0, false, false);
            }
        }
        int code = (jn << 4) + Mloc;
        float en = enorm[code];
#pragma unroll
        for (int i = 0; i < 8; ++i) {
            float s = en - 2.0f * (acc0[i] + acc1[i]);   // ||x||^2 row-const: dropped
            bool better = s < mval[i];
            mval[i] = better ? s : mval[i];
            midx[i] = better ? code : midx[i];
        }
        __syncthreads();   // all waves done reading buf before TDM reuses it
    }

    // Per row: 16 lanes of one half hold its 16 N-slots -> xor-tree min (idx tie-break)
#pragma unroll
    for (int off = 8; off >= 1; off >>= 1) {
#pragma unroll
        for (int i = 0; i < 8; ++i) {
            float ov = __shfl_xor(mval[i], off, 32);
            int   oi = __shfl_xor(midx[i], off, 32);
            bool take = (ov < mval[i]) || (ov == mval[i] && oi < midx[i]);
            mval[i] = take ? ov : mval[i];
            midx[i] = take ? oi : midx[i];
        }
    }
    if (Mloc == 0) {
        int rbase = (blk << 6) + (wave << 4) + ((lane < 16) ? 0 : 8);
#pragma unroll
        for (int i = 0; i < 8; ++i) {
            ind[rbase + i]   = midx[i];
            ind_f[rbase + i] = (float)midx[i];
        }
    }
}

// ---------------------------------------------------------------------------
// quantize_out (B,C,H,W) + diff (B,H,W,C) via LDS transpose, fused embed_sum scatter.
// grid: (cTile=8, hwTile=32, b=64), 256 threads, 32hw x 32c tile.
__global__ __launch_bounds__(256) void k_quant_diff(
    const float* __restrict__ inp, const float* __restrict__ embed,
    const int* __restrict__ ind, float* __restrict__ qout,
    float* __restrict__ diff, float* __restrict__ esum)
{
    __shared__ float T[32][33];
    __shared__ int   I[32];
    int c0  = blockIdx.x << 5;
    int hw0 = blockIdx.y << 5;
    int b   = blockIdx.z;
    int t   = threadIdx.x;
    if (t < 32) I[t] = ind[b * 1024 + hw0 + t];
    __syncthreads();

    int hw  = t & 31;
    int idx = I[hw];
#pragma unroll
    for (int j = 0; j < 4; ++j) {
        int cl = (t >> 5) + 8 * j;
        int c  = c0 + cl;
        size_t goff = (((size_t)b * 256 + c) << 10) + hw0 + hw;
        float x = inp[goff];
        float q = embed[((size_t)c << 10) + idx];   // gather: 1MB table, L2-resident
        qout[goff] = q;
        T[hw][cl] = q - x;
        atomicAdd(&esum[((size_t)c << 10) + idx], x);  // global_atomic_add_f32, no return
    }
    __syncthreads();
#pragma unroll
    for (int j = 0; j < 4; ++j) {
        int hwl = (t >> 5) + 8 * j;
        int c   = t & 31;
        diff[(((size_t)b * 1024 + hw0 + hwl) << 8) + c0 + c] = T[hwl][c];
    }
}

// onehot counts
__global__ void k_count(const int* __restrict__ ind, float* __restrict__ onehot) {
    int n = blockIdx.x * blockDim.x + threadIdx.x;
    if (n < NI) atomicAdd(&onehot[ind[n]], 1.0f);
}

// cluster_size_new + n = sum(cluster_size_new)   (single 1024-thread block)
__global__ __launch_bounds__(1024) void k_cluster(
    const float* __restrict__ cs_in, const float* __restrict__ onehot,
    float* __restrict__ cs_out, float* __restrict__ n_out)
{
    __shared__ float red[1024];
    int j = threadIdx.x;
    float v = cs_in[j] * DECAY + onehot[j] * ONE_MINUS_DECAY;
    cs_out[j] = v;
    red[j] = v;
    __syncthreads();
    for (int s = 512; s > 0; s >>= 1) {
        if (j < s) red[j] += red[j + s];
        __syncthreads();
    }
    if (j == 0) *n_out = red[0];
}

// embed_avg_new + embed_new
__global__ void k_embed_new(
    const float* __restrict__ avg_in, const float* __restrict__ esum,
    const float* __restrict__ cs_in, const float* __restrict__ onehot,
    const float* __restrict__ n_ptr, float* __restrict__ avg_out,
    float* __restrict__ embed_out)
{
    int i = blockIdx.x * blockDim.x + threadIdx.x;   // 0..NE2-1
    if (i >= NE2) return;
    int j = i & 1023;
    float an = avg_in[i] * DECAY + esum[i] * ONE_MINUS_DECAY;
    avg_out[i] = an;
    float nv  = *n_ptr;
    float csn = cs_in[j] * DECAY + onehot[j] * ONE_MINUS_DECAY;
    float cs  = (csn + EPSV) / (nv + 1024.0f * EPSV) * nv;
    embed_out[i] = an / cs;
}

// ---------------------------------------------------------------------------
extern "C" void kernel_launch(void* const* d_in, const int* in_sizes, int n_in,
                              void* d_out, int out_size, void* d_ws, size_t ws_size,
                              hipStream_t stream) {
    const float* inp     = (const float*)d_in[0];   // (64,256,32,32)
    const float* embed   = (const float*)d_in[1];   // (256,1024)
    const float* cs_in   = (const float*)d_in[2];   // (1024,)
    const float* avg_in  = (const float*)d_in[3];   // (256,1024)

    float* out       = (float*)d_out;
    float* qout      = out;                         // 16777216
    float* diff      = out + NQ;                    // 16777216
    float* ind_f     = out + NQ + NDF;              // 65536
    float* embed_new = out + NQ + NDF + NI;         // 262144
    float* cs_new    = embed_new + NE2;             // 1024
    float* avg_new   = cs_new + 1024;               // 262144

    char*   ws     = (char*)d_ws;
    int*    ind    = (int*)ws;                      //  262144 B
    __bf16* ehl    = (__bf16*)(ws + 262144);        // 1048576 B (hi+lo per jn)
    float*  enorm  = (float*)(ws + 1310720);        //    4096 B
    float*  onehot = (float*)(ws + 1314816);        //    4096 B
    float*  esum   = (float*)(ws + 1318912);        // 1048576 B (contiguous after onehot)
    float*  n_ws   = (float*)(ws + 2367488);        //       4 B

    // 1. zero EMA accumulators (onehot + embed_sum contiguous: 263168 floats)
    k_zero<<<(263168 + 255) / 256, 256, 0, stream>>>(onehot, 263168);
    // 2. pre-swizzle codebook into per-N-tile bf16 hi/lo TDM blocks (L2-resident)
    k_prep_embed<<<64, 256, 0, stream>>>(embed, ehl);
    // 3. exact fp32 ||e||^2
    k_enorm<<<4, 256, 0, stream>>>(embed, enorm);
    // 4. fused bf16x3 WMMA distance GEMM + argmin; TDM double-buffered B panel
    k_dist_argmin<<<1024, 128, 0, stream>>>(inp, ehl, enorm, ind, ind_f);
    // 5. quantize + diff (LDS transpose) + embed_sum scatter
    k_quant_diff<<<dim3(8, 32, 64), 256, 0, stream>>>(inp, embed, ind, qout, diff, esum);
    // 6. onehot counts
    k_count<<<NI / 256, 256, 0, stream>>>(ind, onehot);
    // 7. cluster_size EMA + total n
    k_cluster<<<1, 1024, 0, stream>>>(cs_in, onehot, cs_new, n_ws);
    // 8. embed_avg EMA + normalized embed_new
    k_embed_new<<<NE2 / 256, 256, 0, stream>>>(avg_in, esum, cs_in, onehot, n_ws,
                                               avg_new, embed_new);
}
